// SelfAttentionLayer_7464653160729
// MI455X (gfx1250) — compile-verified, matrix operand order checked by reading, not run on hardware
//
#include <hip/hip_runtime.h>
#include <hip/hip_bf16.h>

// ---------------------------------------------------------------------------
// Graph attention layer for MI455X (gfx1250), wave32 + WMMA bf16.
//
// N=10000 nodes, E=320000 edges, D=256, ED=128, H=8 heads (d=32).
// All GEMMs use v_wmma_f32_16x16x32_bf16. The FLOP-dominant edge GEMM
// re = silu(t_ij @ Wre) (21 GFLOP) is fused with the attention logit so the
// 328MB re tensor is never materialized. Per-wave tile is 16x128 (8 accum
// tiles = 64 VGPRs); B fragments use a depth-2 rotation so two load clauses
// stay in flight under the WMMAs.
// ---------------------------------------------------------------------------

#define NN 10000
#define NE 320000
#define DD 256
#define ED 128
#define NH 8

typedef __bf16 bf16_t;
typedef __attribute__((ext_vector_type(16))) __bf16 v16bf;
typedef __attribute__((ext_vector_type(8)))  __bf16 v8bf;
typedef __attribute__((ext_vector_type(8)))  float  v8f;

// ---- WMMA fragment loaders -------------------------------------------------
// 16-bit A/B 16x32 layout (ISA 7.12.2): lane<16 holds K = kt+0..7 then
// kt+16..23; lane>=16 holds K = kt+8..15 then kt+24..31; row/col = lane&15.
static __device__ __forceinline__ v16bf frag_bf16(const bf16_t* __restrict__ base, int sel) {
  v8bf lo = *(const v8bf*)(base + sel);
  v8bf hi = *(const v8bf*)(base + sel + 16);
  v16bf r;
#pragma unroll
  for (int i = 0; i < 8; ++i) { r[i] = lo[i]; r[i + 8] = hi[i]; }
  return r;
}

static __device__ __forceinline__ v16bf frag_f32(const float* __restrict__ base, int sel) {
  v16bf r;
#pragma unroll
  for (int i = 0; i < 8; ++i) {
    r[i]     = (bf16_t)base[sel + i];
    r[i + 8] = (bf16_t)base[sel + 16 + i];
  }
  return r;
}

static __device__ __forceinline__ float silu_f(float x) {
  return x / (1.0f + __expf(-x));
}

// Ordered-float encoding for atomicMax on unsigned
static __device__ __forceinline__ unsigned enc_f(float f) {
  unsigned u = __float_as_uint(f);
  return (u & 0x80000000u) ? ~u : (u | 0x80000000u);
}
static __device__ __forceinline__ float dec_f(unsigned e) {
  return (e & 0x80000000u) ? __uint_as_float(e ^ 0x80000000u) : __uint_as_float(~e);
}
#define ENC_NEG_INF 0x007FFFFFu   // enc_f(-inf)

// ---- LayerNorm: one wave per node, emit bf16 -------------------------------
__global__ void gat_ln_kernel(const float* __restrict__ h,
                              const float* __restrict__ g,
                              const float* __restrict__ b,
                              bf16_t* __restrict__ hn) {
  int wave = (blockIdx.x * blockDim.x + threadIdx.x) >> 5;
  int lane = threadIdx.x & 31;
  if (wave >= NN) return;
  const float* row = h + (size_t)wave * DD;
  float x[8];
  float s = 0.f;
#pragma unroll
  for (int i = 0; i < 8; ++i) { x[i] = row[lane * 8 + i]; s += x[i]; }
#pragma unroll
  for (int o = 16; o > 0; o >>= 1) s += __shfl_xor(s, o, 32);
  float mu = s * (1.0f / DD);
  float vs = 0.f;
#pragma unroll
  for (int i = 0; i < 8; ++i) { float d = x[i] - mu; vs += d * d; }
#pragma unroll
  for (int o = 16; o > 0; o >>= 1) vs += __shfl_xor(vs, o, 32);
  float rs = rsqrtf(vs * (1.0f / DD) + 1e-5f);
#pragma unroll
  for (int i = 0; i < 8; ++i) {
    int c = lane * 8 + i;
    hn[(size_t)wave * DD + c] = (bf16_t)((x[i] - mu) * rs * g[c] + b[c]);
  }
}

// ---- Weight pack: fp32 [K,N] -> bf16 [N,K] (B-operand friendly) ------------
__global__ void gat_pack_kernel(const float* __restrict__ W, bf16_t* __restrict__ WT,
                                int K, int N) {
  int i = blockIdx.x * blockDim.x + threadIdx.x;
  if (i >= K * N) return;
  int k = i / N, n = i - k * N;
  WT[(size_t)n * K + k] = (bf16_t)W[i];
}

// ---- Generic WMMA GEMM: C[M,256] = act(A[M,K] @ B[K,256] + bias) -----------
// One wave per (16-row strip, 128-col group) job: 8 accum tiles = 64 VGPRs.
template <int K, bool SILU, bool BF16OUT>
__global__ void gat_gemm_kernel(const bf16_t* __restrict__ A,
                                const bf16_t* __restrict__ BT,   // [256, K]
                                const float* __restrict__ bias,  // [256] or null
                                void* __restrict__ Cout,
                                int njobs) {
  int job  = (blockIdx.x * blockDim.x + threadIdx.x) >> 5;
  int lane = threadIdx.x & 31;
  if (job >= njobs) return;
  int strip = job >> 1;
  int cg    = job & 1;                  // column group: cols [cg*128, cg*128+128)
  const int sel   = (lane >> 4) << 3;   // 0 or 8
  const int half  = lane & 15;
  const int rbase = sel;

  const bf16_t* arow = A + ((size_t)strip * 16 + half) * K;
  const bf16_t* brow = BT + ((size_t)cg * 128 + half) * K;

  v8f acc[8];
#pragma unroll
  for (int nt = 0; nt < 8; ++nt)
#pragma unroll
    for (int i = 0; i < 8; ++i) acc[nt][i] = 0.0f;

  v16bf af = frag_bf16(arow, sel);
#pragma unroll
  for (int kt = 0; kt < K; kt += 32) {
    // depth-2 B rotation: wmma(nt) consumes the fragment loaded 2 iters ago
    v16bf b0 = frag_bf16(brow + kt, sel);
    v16bf b1 = frag_bf16(brow + (size_t)16 * K + kt, sel);
    v16bf anext;
    if (kt + 32 < K) anext = frag_bf16(arow + kt + 32, sel);
#pragma unroll
    for (int nt = 0; nt < 8; ++nt) {
      v16bf bn;
      if (nt < 6) bn = frag_bf16(brow + (size_t)(nt + 2) * 16 * K + kt, sel);
      acc[nt] = __builtin_amdgcn_wmma_f32_16x16x32_bf16(
          false, af, false, b0, (short)0, acc[nt], false, false);
      b0 = b1;
      b1 = bn;
    }
    af = anext;
  }

#pragma unroll
  for (int nt = 0; nt < 8; ++nt) {
    int col = cg * 128 + nt * 16 + half;
    float bv = bias ? bias[col] : 0.0f;
#pragma unroll
    for (int i = 0; i < 8; ++i) {
      int m = strip * 16 + rbase + i;
      float x = acc[nt][i] + bv;
      if (SILU) x = silu_f(x);
      if (BF16OUT) ((bf16_t*)Cout)[(size_t)m * DD + col] = (bf16_t)x;
      else         ((float*)Cout)[(size_t)m * DD + col] = x;
    }
  }
}

// ---- Init segment-max / segment-sum buffers --------------------------------
__global__ void gat_init_kernel(unsigned* __restrict__ m_enc, float* __restrict__ denom,
                                int n) {
  int i = blockIdx.x * blockDim.x + threadIdx.x;
  if (i < n) { m_enc[i] = ENC_NEG_INF; denom[i] = 0.0f; }
}

// ---- Fused edge kernel: re = silu(t @ Wre) via WMMA, then logits -----------
// Block = 4 waves = 2 edge-strips x 2 column halves (32 edges per block).
#define RSTRIDE 260   // 256 + 4-float pad: avoids 64-bank conflicts on LDS
__global__ void gat_edge_kernel(const float* __restrict__ T,      // [E,128] f32
                                const bf16_t* __restrict__ WreT,  // [256,128] bf16
                                const int* __restrict__ eidx,     // [2,E]
                                const float* __restrict__ q,      // [N,256]
                                const float* __restrict__ k,      // [N,256]
                                float* __restrict__ a_out,        // [E,8]
                                unsigned* __restrict__ m_enc) {   // [N,8]
  __shared__ float re_lds[2 * 16 * RSTRIDE];
  int warp  = threadIdx.x >> 5;   // 0..3
  int lane  = threadIdx.x & 31;
  int sloc  = warp >> 1;          // local strip 0/1
  int cg    = warp & 1;           // column half 0/1
  int e0b   = blockIdx.x * 32;    // first edge of block
  int e0    = e0b + sloc * 16;    // first edge of this wave's strip
  int sel   = (lane >> 4) << 3;
  int half  = lane & 15;
  int rbase = sel;

  // Warm the cache for the index loads used after the GEMM phase.
  __builtin_prefetch(eidx + e0b, 0, 0);        // global_prefetch_b8
  __builtin_prefetch(eidx + NE + e0b, 0, 0);

  const float*  arow = T + (size_t)(e0 + half) * ED;
  const bf16_t* brow = WreT + ((size_t)cg * 128 + half) * ED;

  v8f acc[8];
#pragma unroll
  for (int nt = 0; nt < 8; ++nt)
#pragma unroll
    for (int i = 0; i < 8; ++i) acc[nt][i] = 0.0f;

  v16bf af = frag_f32(arow, sel);          // f32 -> bf16 on the fly
#pragma unroll
  for (int kt = 0; kt < ED; kt += 32) {
    v16bf b0 = frag_bf16(brow + kt, sel);
    v16bf b1 = frag_bf16(brow + (size_t)16 * ED + kt, sel);
    v16bf anext;
    if (kt + 32 < ED) anext = frag_f32(arow + kt + 32, sel);
#pragma unroll
    for (int nt = 0; nt < 8; ++nt) {
      v16bf bn;
      if (nt < 6) bn = frag_bf16(brow + (size_t)(nt + 2) * 16 * ED + kt, sel);
      acc[nt] = __builtin_amdgcn_wmma_f32_16x16x32_bf16(
          false, af, false, b0, (short)0, acc[nt], false, false);
      b0 = b1;
      b1 = bn;
    }
    af = anext;
  }

  // SiLU + park this wave's 16x128 quarter of the re tile in LDS
  float* myl = re_lds + sloc * 16 * RSTRIDE;
#pragma unroll
  for (int nt = 0; nt < 8; ++nt)
#pragma unroll
    for (int i = 0; i < 8; ++i)
      myl[(rbase + i) * RSTRIDE + cg * 128 + nt * 16 + half] = silu_f(acc[nt][i]);

  __syncthreads();

  // 32 edges x 8 heads = 256 logits per block; 2 per thread.
  // a = sum_d q[n_i][h*32+d] * k[n_j][h*32+d] * re[e][h*32+d]
  int tid = threadIdx.x;
#pragma unroll
  for (int rr = 0; rr < 2; ++rr) {
    int idx = tid + 128 * rr;       // 0..255
    int sl  = idx >> 7;             // strip 0/1
    int eh  = idx & 127;
    int el  = eh >> 3, hd = eh & 7;
    int e   = e0b + sl * 16 + el;
    int nj  = eidx[e];              // source
    int ni  = eidx[NE + e];         // destination
    const float4* qp = (const float4*)(q + (size_t)ni * DD + hd * 32);
    const float4* kp = (const float4*)(k + (size_t)nj * DD + hd * 32);
    const float*  rp = re_lds + sl * 16 * RSTRIDE + el * RSTRIDE + hd * 32;
    float dot = 0.0f;
#pragma unroll
    for (int c = 0; c < 8; ++c) {
      float4 qv = qp[c], kv = kp[c];
      dot += qv.x * kv.x * rp[c * 4 + 0];
      dot += qv.y * kv.y * rp[c * 4 + 1];
      dot += qv.z * kv.z * rp[c * 4 + 2];
      dot += qv.w * kv.w * rp[c * 4 + 3];
    }
    a_out[(size_t)e * NH + hd] = dot;
    atomicMax(&m_enc[(unsigned)ni * NH + hd], enc_f(dot));
  }
}

// ---- exp(a - max) + segment-sum of denominator -----------------------------
__global__ void gat_softmax_num_kernel(const float* __restrict__ a,
                                       const unsigned* __restrict__ m_enc,
                                       const int* __restrict__ eidx,
                                       float* __restrict__ ex,
                                       float* __restrict__ denom) {
  int i = blockIdx.x * blockDim.x + threadIdx.x;
  if (i >= NE * NH) return;
  int e = i >> 3, hd = i & 7;
  int ni = eidx[NE + e];
  float v = __expf(a[i] - dec_f(m_enc[ni * NH + hd]));
  ex[i] = v;
  atomicAdd(&denom[ni * NH + hd], v);
}

// ---- out[e, h*32+d] = (ex / denom[n_i,h]) * v[n_j, h*32+d] -----------------
__global__ void gat_output_kernel(const float* __restrict__ ex,
                                  const float* __restrict__ denom,
                                  const int* __restrict__ eidx,
                                  const float* __restrict__ v,
                                  float* __restrict__ out) {
  int i = blockIdx.x * blockDim.x + threadIdx.x;   // one (edge, head) pair
  if (i >= NE * NH) return;
  int e = i >> 3, hd = i & 7;
  int ni = eidx[NE + e];
  int nj = eidx[e];
  float alpha = ex[i] / denom[ni * NH + hd];
  const float4* vp = (const float4*)(v + (size_t)nj * DD + hd * 32);
  float4* op = (float4*)(out + (size_t)e * DD + hd * 32);
#pragma unroll
  for (int c = 0; c < 8; ++c) {
    float4 t = vp[c];
    float4 o;
    o.x = alpha * t.x; o.y = alpha * t.y; o.z = alpha * t.z; o.w = alpha * t.w;
    op[c] = o;
  }
}

// ---------------------------------------------------------------------------
extern "C" void kernel_launch(void* const* d_in, const int* in_sizes, int n_in,
                              void* d_out, int out_size, void* d_ws, size_t ws_size,
                              hipStream_t stream) {
  (void)in_sizes; (void)n_in; (void)out_size; (void)ws_size;
  const float* h    = (const float*)d_in[0];
  const float* t_ij = (const float*)d_in[1];
  const int*   eidx = (const int*)d_in[2];
  const float* ln_g = (const float*)d_in[3];
  const float* ln_b = (const float*)d_in[4];
  const float* Wq   = (const float*)d_in[5];
  const float* Wk   = (const float*)d_in[6];
  const float* Wre  = (const float*)d_in[7];
  const float* W1   = (const float*)d_in[8];
  const float* b1   = (const float*)d_in[9];
  const float* W2   = (const float*)d_in[10];
  const float* b2   = (const float*)d_in[11];
  float* out = (float*)d_out;

  // Workspace carve-out (~63 MB)
  char* ws = (char*)d_ws;
  size_t off = 0;
  auto alloc = [&](size_t bytes) -> void* {
    void* p = ws + off;
    off += (bytes + 255) & ~(size_t)255;
    return p;
  };
  bf16_t* hn_bf  = (bf16_t*)alloc((size_t)NN * DD * 2);
  bf16_t* WqT    = (bf16_t*)alloc((size_t)DD * DD * 2);
  bf16_t* WkT    = (bf16_t*)alloc((size_t)DD * DD * 2);
  bf16_t* W1T    = (bf16_t*)alloc((size_t)DD * DD * 2);
  bf16_t* W2T    = (bf16_t*)alloc((size_t)DD * DD * 2);
  bf16_t* WreT   = (bf16_t*)alloc((size_t)DD * ED * 2);
  bf16_t* hmid   = (bf16_t*)alloc((size_t)NN * DD * 2);
  float*  qbuf   = (float*)alloc((size_t)NN * DD * 4);
  float*  kbuf   = (float*)alloc((size_t)NN * DD * 4);
  float*  vbuf   = (float*)alloc((size_t)NN * DD * 4);
  float*  abuf   = (float*)alloc((size_t)NE * NH * 4);
  float*  exbuf  = (float*)alloc((size_t)NE * NH * 4);
  unsigned* menc = (unsigned*)alloc((size_t)NN * NH * 4);
  float*  denom  = (float*)alloc((size_t)NN * NH * 4);

  // 1) LayerNorm -> bf16 hn
  gat_ln_kernel<<<(NN + 7) / 8, 256, 0, stream>>>(h, ln_g, ln_b, hn_bf);

  // 2) Pack weights to bf16 [N,K]
  gat_pack_kernel<<<(DD * DD + 255) / 256, 256, 0, stream>>>(Wq,  WqT,  DD, DD);
  gat_pack_kernel<<<(DD * DD + 255) / 256, 256, 0, stream>>>(Wk,  WkT,  DD, DD);
  gat_pack_kernel<<<(DD * DD + 255) / 256, 256, 0, stream>>>(W1,  W1T,  DD, DD);
  gat_pack_kernel<<<(DD * DD + 255) / 256, 256, 0, stream>>>(W2,  W2T,  DD, DD);
  gat_pack_kernel<<<(ED * DD + 255) / 256, 256, 0, stream>>>(Wre, WreT, ED, DD);

  // 3) Node GEMMs via WMMA: 625 strips x 2 col-groups = 1250 wave-jobs
  const int njobs  = (NN / 16) * 2;            // 1250
  const int gblk   = (njobs + 7) / 8;          // 157 blocks of 8 waves
  gat_gemm_kernel<DD, false, false><<<gblk, 256, 0, stream>>>(hn_bf, WqT, nullptr, qbuf, njobs);
  gat_gemm_kernel<DD, false, false><<<gblk, 256, 0, stream>>>(hn_bf, WkT, nullptr, kbuf, njobs);
  gat_gemm_kernel<DD, true,  true ><<<gblk, 256, 0, stream>>>(hn_bf, W1T, b1, hmid, njobs);
  gat_gemm_kernel<DD, false, false><<<gblk, 256, 0, stream>>>(hmid,  W2T, b2, vbuf, njobs);

  // 4) Init softmax scratch
  gat_init_kernel<<<(NN * NH + 255) / 256, 256, 0, stream>>>(menc, denom, NN * NH);

  // 5) Fused edge GEMM (WMMA) + attention logits + segment max
  gat_edge_kernel<<<NE / 32, 128, 0, stream>>>(t_ij, WreT, eidx, qbuf, kbuf, abuf, menc);

  // 6) exp + segment sum
  gat_softmax_num_kernel<<<(NE * NH + 255) / 256, 256, 0, stream>>>(abuf, menc, eidx, exbuf, denom);

  // 7) normalize + scale v_j -> output
  gat_output_kernel<<<(NE * NH + 255) / 256, 256, 0, stream>>>(exbuf, denom, eidx, vbuf, out);
}
